// Attention_75290776699348
// MI455X (gfx1250) — matrix operand order repaired and checked by reading, not compiled, not run on hardware
//
#include <hip/hip_runtime.h>

#define S_LEN 4096
#define D_DIM 1024
#define NHEAD 16
#define HDIM  64

// Padded LDS row strides (f16 elems). TDM appends 16B pad per row so 16
// reading lanes land on distinct LDS banks for ds_load_b128.
#define WROW 72   // 64 f16 row + 8  (GEMM W tile, K tile)
#define VROW 40   // 32 f16 row + 8  (V tile)

typedef __attribute__((ext_vector_type(16))) _Float16 v16h;
typedef __attribute__((ext_vector_type(8)))  _Float16 v8h;
typedef __attribute__((ext_vector_type(8)))  float    v8f;
typedef __attribute__((ext_vector_type(4)))  unsigned int u32x4;
typedef __attribute__((ext_vector_type(8)))  int i32x8;
typedef __attribute__((ext_vector_type(4)))  int i32x4;

union Frag16 { v8h h[2]; v16h v; };

#if defined(__has_builtin)
#if __has_builtin(__builtin_amdgcn_tensor_load_to_lds)
#define USE_TDM 1
#endif
#endif
#ifndef USE_TDM
#define USE_TDM 0
#endif

// ---------------------------------------------------------------------------
// fp32 -> fp16 elementwise convert
// ---------------------------------------------------------------------------
__global__ void cvt_f32_to_f16(const float* __restrict__ in,
                               _Float16* __restrict__ out, int n) {
  int i = blockIdx.x * blockDim.x + threadIdx.x;
  if (i < n) out[i] = (_Float16)in[i];
}

#if USE_TDM
// TDM load of a (tile_d1 rows x tile_d0 f16) tile from a row-major matrix
// (row length row_len f16, tensor_rows rows) into LDS at byte offset ldsOff.
// 16B of padding is appended per row: pad_code = pad_interval code for one
// row (2^(code+1) dwords), pad_amount code 3 = 4 dwords.
// D# packed per CDNA5 ISA 8.3/8.4.
__device__ inline void tdm_load_tile(const _Float16* gptr, unsigned ldsOff,
                                     int tile_d0, int tile_d1, int row_len,
                                     int tensor_rows, int pad_code) {
  unsigned long long ga = (unsigned long long)(size_t)gptr;
  u32x4 g0;
  g0[0] = 1u;                                     // count=1, user descriptor
  g0[1] = ldsOff;                                 // lds_addr [63:32]
  g0[2] = (unsigned)(ga & 0xFFFFFFFFu);           // global_addr lo
  g0[3] = (unsigned)((ga >> 32) & 0x01FFFFFFu)    // global_addr hi [120:64]
          | (2u << 30);                           // type=2 (image) [127:126]
  i32x8 g1;
  g1[0] = (1 << 16)            // data_size=1 -> 2 bytes/elem  [17:16]
        | (1 << 20)            // pad_enable                   [20]
        | (pad_code << 22)     // pad_interval                 [24:22]
        | (3 << 25);           // pad_amount: 4 dwords (16B)   [31:25]
  g1[1] = (int)((row_len & 0xFFFF) << 16);        // tensor_dim0 lo16 @ [63:48]
  g1[2] = (int)(((row_len >> 16) & 0xFFFF) |      // tensor_dim0 hi16 @ [79:64]
                ((tensor_rows & 0xFFFF) << 16));  // tensor_dim1 lo16 @ [95:80]
  g1[3] = (int)(((tensor_rows >> 16) & 0xFFFF) |  // tensor_dim1 hi16 @ [111:96]
                ((tile_d0 & 0xFFFF) << 16));      // tile_dim0 @ [127:112]
  g1[4] = (int)(tile_d1 & 0xFFFF);                // tile_dim1 @ [143:128]
  g1[5] = row_len;                                // tensor_dim0_stride lo32
  g1[6] = 0;
  g1[7] = 0;
  i32x4 z4 = {0, 0, 0, 0};
#if defined(__clang_major__) && __clang_major__ >= 23
  i32x8 z8 = {};
  __builtin_amdgcn_tensor_load_to_lds(g0, g1, z4, z4, z8, 0);
#else
  __builtin_amdgcn_tensor_load_to_lds(g0, g1, z4, z4, 0);
#endif
}
#endif

// ---------------------------------------------------------------------------
// y = A(4096x1024 f16 row-major) @ W(1024x1024 f16 row-major, N x K)^T
// Block = 8 waves = 128 M-rows x 64 N-cols. 64x64 B tile staged in LDS by the
// Tensor Data Mover (padded rows, conflict-free reads), double-buffered one
// stage ahead; 8 WMMAs per barrier pair. A fragments register double-buffered.
// mode 0: Q [H][S][64] f16 *0.125 | 1: K [H][S][64] | 2: V^T [H][64][S]
// mode 3: fp32 [S][D] to outf
// ---------------------------------------------------------------------------
__global__ __launch_bounds__(256)
void gemm_xwt_wmma(const _Float16* __restrict__ A,
                   const _Float16* __restrict__ W,
                   _Float16* __restrict__ outh,
                   float* __restrict__ outf,
                   int mode) {
  __shared__ __align__(16) _Float16 Bsh[2][64 * WROW];  // 2 x 9KB
  const int lane = threadIdx.x & 31;
  const int wave = threadIdx.x >> 5;
  const int mblk   = blockIdx.x & 31;   // 32 M-blocks of 128 rows
  const int nstrip = blockIdx.x >> 5;   // 16 N strips of 64 cols
  const int m0 = mblk * 128 + wave * 16;
  const int n0 = nstrip * 64;
  const int half = lane >> 4;
  const int l15  = lane & 15;

  const _Float16* wtile = W + (size_t)n0 * D_DIM;       // row n0, col 0
  const _Float16* arow  = A + (size_t)(m0 + l15) * D_DIM;

#if USE_TDM
  if (wave == 0)
    tdm_load_tile(wtile, (unsigned)(size_t)&Bsh[0][0], 64, 64, D_DIM, D_DIM, 4);
#endif

  v8f acc[4] = {};
  // A fragments for stage 0 (k = 0..63)
  v8h aa0 = *(const v8h*)(arow + half * 8);
  v8h aa1 = *(const v8h*)(arow + 16 + half * 8);
  v8h aa2 = *(const v8h*)(arow + 32 + half * 8);
  v8h aa3 = *(const v8h*)(arow + 48 + half * 8);

  const int NSTAGE = D_DIM / 64;  // 16
  for (int s = 0; s < NSTAGE; ++s) {
    const int kb = s * 64;
    const bool more = (s + 1) < NSTAGE;

    __syncthreads();  // everyone done reading buffer (s+1)&1 from stage s-1
#if USE_TDM
    if (wave == 0) {
      if (more) {
        tdm_load_tile(wtile + kb + 64, (unsigned)(size_t)&Bsh[(s + 1) & 1][0],
                      64, 64, D_DIM, D_DIM, 4);
        __builtin_amdgcn_s_wait_tensorcnt(1);  // buf s&1 ready, next in flight
      } else {
        __builtin_amdgcn_s_wait_tensorcnt(0);
      }
    }
    __syncthreads();
#else
    {  // cooperative fallback: 256 thr x 16 f16 = 64x64 tile, padded rows
      const int row = threadIdx.x >> 2;
      const int c   = (threadIdx.x & 3) * 16;
      *(v8h*)&Bsh[s & 1][row * WROW + c] =
          *(const v8h*)(wtile + (size_t)row * D_DIM + kb + c);
      *(v8h*)&Bsh[s & 1][row * WROW + c + 8] =
          *(const v8h*)(wtile + (size_t)row * D_DIM + kb + c + 8);
    }
    __syncthreads();
#endif

    // prefetch next stage's A (first use is next stage)
    v8h na0 = aa0, na1 = aa1, na2 = aa2, na3 = aa3;
    if (more) {
      na0 = *(const v8h*)(arow + kb + 64 + half * 8);
      na1 = *(const v8h*)(arow + kb + 80 + half * 8);
      na2 = *(const v8h*)(arow + kb + 96 + half * 8);
      na3 = *(const v8h*)(arow + kb + 112 + half * 8);
    }

    const _Float16* bbase = &Bsh[s & 1][0];
#pragma unroll
    for (int sub = 0; sub < 2; ++sub) {
      Frag16 bf[4];
#pragma unroll
      for (int t = 0; t < 4; ++t) {
        const _Float16* bp =
            bbase + (t * 16 + l15) * WROW + sub * 32 + half * 8;
        bf[t].h[0] = *(const v8h*)(bp);
        bf[t].h[1] = *(const v8h*)(bp + 16);
      }
      Frag16 af;
      af.h[0] = sub ? aa2 : aa0;
      af.h[1] = sub ? aa3 : aa1;
#pragma unroll
      for (int t = 0; t < 4; ++t) {
        acc[t] = __builtin_amdgcn_wmma_f32_16x16x32_f16(
            false, af.v, false, bf[t].v, (short)0, acc[t], false, false);
      }
    }
    aa0 = na0; aa1 = na1; aa2 = na2; aa3 = na3;
  }

#pragma unroll
  for (int t = 0; t < 4; ++t) {
    const int n    = n0 + t * 16 + l15;
    const int head = n >> 6;
    const int hd   = n & 63;
#pragma unroll
    for (int r = 0; r < 8; ++r) {
      const int m = m0 + r + 8 * half;
      const float v = acc[t][r];
      if (mode == 0)
        outh[(size_t)head * S_LEN * HDIM + (size_t)m * HDIM + hd] =
            (_Float16)(v * 0.125f);
      else if (mode == 1)
        outh[(size_t)head * S_LEN * HDIM + (size_t)m * HDIM + hd] = (_Float16)v;
      else if (mode == 2)
        outh[(size_t)head * HDIM * S_LEN + (size_t)hd * S_LEN + m] = (_Float16)v;
      else
        outf[(size_t)m * D_DIM + n] = v;
    }
  }
}

// ---------------------------------------------------------------------------
// Causal flash attention, block-cooperative K/V staging.
// Block = one head x 128 q rows; wave w owns q rows [q0+16w, q0+16w+16).
// Per 32-key chunk, wave 0 TDM-loads the K tile (32x64) and V tile (64x32,
// from V^T) into double-buffered, bank-padded LDS shared by all 8 waves
// (8x less L2 traffic than per-wave streaming). Tail waves past their causal
// range skip compute but keep the uniform barrier schedule.
// Q,K: [H][S][64] f16 (Q pre-scaled by 1/8).  Vt: [H][64][S] f16.
// ---------------------------------------------------------------------------
__global__ __launch_bounds__(256)
void flash_attn_wmma(const _Float16* __restrict__ Q,
                     const _Float16* __restrict__ K,
                     const _Float16* __restrict__ Vt,
                     _Float16* __restrict__ ctx) {
  __shared__ __align__(16) _Float16 Ksh[2][32 * WROW];  // 2 x 4.5KB
  __shared__ __align__(16) _Float16 Vsh[2][64 * VROW];  // 2 x 5KB
  __shared__ __align__(16) _Float16 plds[8 * 16 * 32];  // P bounce, 1KB/wave
  const int lane = threadIdx.x & 31;
  const int wave = threadIdx.x >> 5;
  _Float16* my = plds + wave * 16 * 32;

  const int head = blockIdx.x >> 5;   // 16 heads
  const int qblk = blockIdx.x & 31;   // 32 blocks of 128 q rows
  const int q0    = qblk * 128;
  const int qbase = q0 + wave * 16;
  const int half  = lane >> 4;
  const int l15   = lane & 15;

  const _Float16* Qh = Q + (size_t)head * S_LEN * HDIM;
  const _Float16* Kh = K + (size_t)head * S_LEN * HDIM;
  const _Float16* Vh = Vt + (size_t)head * HDIM * S_LEN;

  // Q A-fragments for d=0..31 and d=32..63 (row = qbase + l15)
  const _Float16* qrow = Qh + (size_t)(qbase + l15) * HDIM;
  Frag16 qa0, qa1;
  qa0.h[0] = *(const v8h*)(qrow + half * 16);
  qa0.h[1] = *(const v8h*)(qrow + half * 16 + 8);
  qa1.h[0] = *(const v8h*)(qrow + 32 + half * 16);
  qa1.h[1] = *(const v8h*)(qrow + 32 + half * 16 + 8);

  v8f o[4] = {};
  float mrow[8], lrow[8];
#pragma unroll
  for (int r = 0; r < 8; ++r) { mrow[r] = -1e30f; lrow[r] = 0.0f; }

  const int my_jend    = qbase + 16;  // this wave's causal range (exclusive)
  const int block_jend = q0 + 128;    // block-wide range

#if USE_TDM
  if (wave == 0) {
    tdm_load_tile(Kh, (unsigned)(size_t)&Ksh[0][0], 64, 32, HDIM, S_LEN, 4);
    tdm_load_tile(Vh, (unsigned)(size_t)&Vsh[0][0], 32, 64, S_LEN, HDIM, 3);
  }
#endif

  int ic = 0;
  for (int jb = 0; jb < block_jend; jb += 32, ++ic) {
    const bool more = (jb + 32) < block_jend;
    const int buf = ic & 1;

    __syncthreads();  // all waves done reading buffer buf^1 from chunk ic-1
#if USE_TDM
    if (wave == 0) {
      if (more) {
        tdm_load_tile(Kh + (size_t)(jb + 32) * HDIM,
                      (unsigned)(size_t)&Ksh[buf ^ 1][0], 64, 32, HDIM, S_LEN, 4);
        tdm_load_tile(Vh + jb + 32,
                      (unsigned)(size_t)&Vsh[buf ^ 1][0], 32, 64, S_LEN, HDIM, 3);
        __builtin_amdgcn_s_wait_tensorcnt(2);  // current pair done, next pair in flight
      } else {
        __builtin_amdgcn_s_wait_tensorcnt(0);
      }
    }
    __syncthreads();
#else
    {  // cooperative fallback, padded layouts
      const int kr = threadIdx.x >> 3, kc = (threadIdx.x & 7) * 8;
      *(v8h*)&Ksh[buf][kr * WROW + kc] =
          *(const v8h*)(Kh + (size_t)(jb + kr) * HDIM + kc);
      const int vr = threadIdx.x >> 2, vc = (threadIdx.x & 3) * 8;
      *(v8h*)&Vsh[buf][vr * VROW + vc] =
          *(const v8h*)(Vh + (size_t)vr * S_LEN + jb + vc);
    }
    __syncthreads();
#endif

    if (jb < my_jend) {
      // ---- K / V fragments from shared LDS tiles ----
      Frag16 kA0, kA1, kB0, kB1, vb[4];
      {
        const _Float16* kp0 = &Ksh[buf][l15 * WROW + half * 16];
        const _Float16* kp1 = &Ksh[buf][(16 + l15) * WROW + half * 16];
        kA0.h[0] = *(const v8h*)(kp0);
        kA0.h[1] = *(const v8h*)(kp0 + 8);
        kA1.h[0] = *(const v8h*)(kp0 + 32);
        kA1.h[1] = *(const v8h*)(kp0 + 40);
        kB0.h[0] = *(const v8h*)(kp1);
        kB0.h[1] = *(const v8h*)(kp1 + 8);
        kB1.h[0] = *(const v8h*)(kp1 + 32);
        kB1.h[1] = *(const v8h*)(kp1 + 40);
      }
#pragma unroll
      for (int dt = 0; dt < 4; ++dt) {
        const _Float16* vp = &Vsh[buf][(dt * 16 + l15) * VROW + half * 16];
        vb[dt].h[0] = *(const v8h*)(vp);
        vb[dt].h[1] = *(const v8h*)(vp + 8);
      }

      // ---- scores: two 16x16 fragments (j tiles jb, jb+16), d reduced 64 ----
      v8f s0 = {}, s1 = {};
      s0 = __builtin_amdgcn_wmma_f32_16x16x32_f16(false, qa0.v, false, kA0.v,
                                                  (short)0, s0, false, false);
      s0 = __builtin_amdgcn_wmma_f32_16x16x32_f16(false, qa1.v, false, kA1.v,
                                                  (short)0, s0, false, false);
      s1 = __builtin_amdgcn_wmma_f32_16x16x32_f16(false, qa0.v, false, kB0.v,
                                                  (short)0, s1, false, false);
      s1 = __builtin_amdgcn_wmma_f32_16x16x32_f16(false, qa1.v, false, kB1.v,
                                                  (short)0, s1, false, false);

      // ---- causal mask (only diagonal-straddling chunks) ----
      if (jb + 31 > qbase) {
        const int j0 = jb + l15, j1 = jb + 16 + l15;
#pragma unroll
        for (int r = 0; r < 8; ++r) {
          const int q = qbase + r + 8 * half;
          if (j0 > q) s0[r] = -1e30f;
          if (j1 > q) s1[r] = -1e30f;
        }
      }

      // ---- online softmax (fp32); row reductions across 16 column lanes ----
      float t[8];
#pragma unroll
      for (int r = 0; r < 8; ++r) t[r] = fmaxf(s0[r], s1[r]);
#pragma unroll
      for (int msk = 1; msk <= 8; msk <<= 1)
#pragma unroll
        for (int r = 0; r < 8; ++r)
          t[r] = fmaxf(t[r], __shfl_xor(t[r], msk, 32));

      float alpha[8];
#pragma unroll
      for (int r = 0; r < 8; ++r) {
        const float mn = fmaxf(mrow[r], t[r]);
        alpha[r] = __expf(mrow[r] - mn);
        mrow[r] = mn;
      }
#pragma unroll
      for (int r = 0; r < 8; ++r) {
        s0[r] = __expf(s0[r] - mrow[r]);
        s1[r] = __expf(s1[r] - mrow[r]);
      }
      float u[8];
#pragma unroll
      for (int r = 0; r < 8; ++r) u[r] = s0[r] + s1[r];
#pragma unroll
      for (int msk = 1; msk <= 8; msk <<= 1)
#pragma unroll
        for (int r = 0; r < 8; ++r) u[r] += __shfl_xor(u[r], msk, 32);
#pragma unroll
      for (int r = 0; r < 8; ++r) lrow[r] = lrow[r] * alpha[r] + u[r];

      // ---- C-layout P -> A-layout fp16 via private LDS bounce ----
#pragma unroll
      for (int r = 0; r < 8; ++r) {
        const int row = r + 8 * half;
        my[row * 32 + l15]      = (_Float16)s0[r];
        my[row * 32 + 16 + l15] = (_Float16)s1[r];
      }
      asm volatile("s_wait_dscnt 0" ::: "memory");  // wave-local DS ordering
      Frag16 pa;
      pa.h[0] = *(const v8h*)(my + l15 * 32 + half * 16);
      pa.h[1] = *(const v8h*)(my + l15 * 32 + half * 16 + 8);

      // ---- rescale running ctx, accumulate P @ V ----
#pragma unroll
      for (int dt = 0; dt < 4; ++dt) {
#pragma unroll
        for (int r = 0; r < 8; ++r) o[dt][r] *= alpha[r];
        o[dt] = __builtin_amdgcn_wmma_f32_16x16x32_f16(
            false, pa.v, false, vb[dt].v, (short)0, o[dt], false, false);
      }
    }
  }

  // ---- normalize and store ctx [S][D] f16 ----
#pragma unroll
  for (int dt = 0; dt < 4; ++dt) {
#pragma unroll
    for (int r = 0; r < 8; ++r) {
      const int m = qbase + r + 8 * half;
      const int d = dt * 16 + l15;
      ctx[(size_t)m * D_DIM + head * HDIM + d] =
          (_Float16)(o[dt][r] / lrow[r]);
    }
  }
}

// ---------------------------------------------------------------------------
// Host-side orchestration
// ---------------------------------------------------------------------------
extern "C" void kernel_launch(void* const* d_in, const int* in_sizes, int n_in,
                              void* d_out, int out_size, void* d_ws,
                              size_t ws_size, hipStream_t stream) {
  const float* x  = (const float*)d_in[0];
  const float* Wq = (const float*)d_in[1];
  const float* Wk = (const float*)d_in[2];
  const float* Wv = (const float*)d_in[3];
  const float* Wo = (const float*)d_in[4];
  float* out = (float*)d_out;

  const size_t NX = (size_t)S_LEN * D_DIM;  // 4M elems
  const size_t NW = (size_t)D_DIM * D_DIM;  // 1M elems

  _Float16* ws  = (_Float16*)d_ws;
  _Float16* xh  = ws;
  _Float16* wqh = xh + NX;
  _Float16* wkh = wqh + NW;
  _Float16* wvh = wkh + NW;
  _Float16* woh = wvh + NW;
  _Float16* Qb  = woh + NW;  // [H][S][64]
  _Float16* Kb  = Qb + NX;   // [H][S][64]
  _Float16* Vtb = Kb + NX;   // [H][64][S]
  _Float16* ctx = Vtb + NX;  // [S][D]

  cvt_f32_to_f16<<<(int)((NX + 255) / 256), 256, 0, stream>>>(x, xh, (int)NX);
  cvt_f32_to_f16<<<(int)((NW + 255) / 256), 256, 0, stream>>>(Wq, wqh, (int)NW);
  cvt_f32_to_f16<<<(int)((NW + 255) / 256), 256, 0, stream>>>(Wk, wkh, (int)NW);
  cvt_f32_to_f16<<<(int)((NW + 255) / 256), 256, 0, stream>>>(Wv, wvh, (int)NW);
  cvt_f32_to_f16<<<(int)((NW + 255) / 256), 256, 0, stream>>>(Wo, woh, (int)NW);

  // 32 M-blocks x 16 N-strips = 512 blocks, 256 threads
  gemm_xwt_wmma<<<512, 256, 0, stream>>>(xh, wqh, Qb, nullptr, 0);
  gemm_xwt_wmma<<<512, 256, 0, stream>>>(xh, wkh, Kb, nullptr, 1);
  gemm_xwt_wmma<<<512, 256, 0, stream>>>(xh, wvh, Vtb, nullptr, 2);

  // 16 heads x 32 q-blocks of 128 rows = 512 blocks
  flash_attn_wmma<<<512, 256, 0, stream>>>(Qb, Kb, Vtb, ctx);

  gemm_xwt_wmma<<<512, 256, 0, stream>>>(ctx, woh, nullptr, out, 3);
}
